// MultiHeadAttention_65850438582896
// MI455X (gfx1250) — matrix operand order repaired and checked by reading, not compiled, not run on hardware
//
#include <hip/hip_runtime.h>
#include <hip/hip_bf16.h>

typedef _Float16 h16;
typedef __attribute__((ext_vector_type(16))) _Float16 v16h;
typedef __attribute__((ext_vector_type(8)))  _Float16 v8h;
typedef __attribute__((ext_vector_type(8)))  float    v8f;

#define BATCH  2
#define SEQL   2048
#define DMODEL 1024
#define HEADS  16
#define HDIM   64

__device__ __forceinline__ v16h mk_frag(v8h lo, v8h hi) {
  v16h r;
#pragma unroll
  for (int i = 0; i < 8; ++i) { r[i] = lo[i]; r[i + 8] = hi[i]; }
  return r;
}

__device__ __forceinline__ v8f wmma_f16(v16h a, v16h b, v8f c) {
  // D = A(16x32 f16) * B(32x16 f16) + C(16x16 f32)
  return __builtin_amdgcn_wmma_f32_16x16x32_f16(false, a, false, b,
                                                (short)0, c, false, false);
}

// CDNA5 async global->LDS copy (ASYNCcnt path). ldsOff = wave-relative LDS byte
// offset (low 32 bits of a generic __shared__ pointer); gp = global address.
// INST_OFFSET applies to BOTH the LDS and global addresses (ISA 08 §4.4).
__device__ __forceinline__ void async_g2l_b128(unsigned ldsOff, const void* gp) {
  asm volatile("global_load_async_to_lds_b128 %0, %1, off"
               :: "v"(ldsOff), "v"(gp) : "memory");
}
__device__ __forceinline__ void async_g2l_b128_off16(unsigned ldsOff, const void* gp) {
  asm volatile("global_load_async_to_lds_b128 %0, %1, off offset:16"
               :: "v"(ldsOff), "v"(gp) : "memory");
}
__device__ __forceinline__ void wait_async0() {
  asm volatile("s_wait_asynccnt 0" ::: "memory");
}

// ---------------------------------------------------------------- cast f32->f16
__global__ void cast_f32_f16(const float* __restrict__ s, h16* __restrict__ d, int n) {
  int i = blockIdx.x * 256 + threadIdx.x;
  if (i < n) d[i] = (h16)s[i];
}

// ---------------------------------------------------------------- tiled WMMA GEMM
// C[M,N] = A[M,K](f16,row-major) * B[K,N](f16,row-major) + bias[N]
// block tile 128x128, 8 waves (4m x 2n), wave tile 32x64 (2x4 16x16 tiles), K-step 32.
// A tile staged via GLOBAL_LOAD_ASYNC_TO_LDS_B128 (ASYNCcnt); B transposed via VGPRs.
template <bool F16OUT>
__global__ __launch_bounds__(256) void gemm_bias(
    const h16* __restrict__ A, const h16* __restrict__ B,
    const float* __restrict__ bias, void* __restrict__ Cout,
    int M, int N, int K) {
  __shared__ h16 lA[128 * 32];   // [row][k]
  __shared__ h16 lBt[128 * 32];  // [n][k]  (B transposed for K-contig fragments)

  const int tid  = threadIdx.x;
  const int lane = tid & 31;
  const int wid  = tid >> 5;
  const int wm   = wid >> 1;     // 0..3
  const int wn   = wid & 1;      // 0..1
  const int hm   = lane & 15;    // 0..15
  const int g    = lane >> 4;    // 0..1
  const int r0   = blockIdx.y * 128;
  const int c0   = blockIdx.x * 128;

  v8f acc[2][4];
#pragma unroll
  for (int i = 0; i < 2; ++i)
#pragma unroll
    for (int j = 0; j < 4; ++j) acc[i][j] = {};

  // A staging: 4096 halves / 256 thr = 16 contiguous halves each (2 x b128 async)
  const int aStart = tid * 16;
  const int aRow   = aStart >> 5;
  const int aCol   = aStart & 31;
  const unsigned ldsA = (unsigned)(size_t)(void*)&lA[aStart];  // LDS byte offset
  // B staging (transpose into lBt): thread -> (k = tid/8, n0 = (tid%8)*16)
  const int bK = tid >> 3;
  const int bN = (tid & 7) * 16;

  for (int k0 = 0; k0 < K; k0 += 32) {
    __syncthreads();
    const h16* gpA = A + (size_t)(r0 + aRow) * K + k0 + aCol;
    async_g2l_b128(ldsA, gpA);
    async_g2l_b128_off16(ldsA, gpA);
    const h16* gpB = B + (size_t)(k0 + bK) * N + c0 + bN;
    {
      v8h x0 = *(const v8h*)gpB;
      v8h x1 = *(const v8h*)(gpB + 8);
#pragma unroll
      for (int i = 0; i < 8; ++i) {
        lBt[(bN + i) * 32 + bK]     = x0[i];
        lBt[(bN + 8 + i) * 32 + bK] = x1[i];
      }
    }
    if (k0 + 32 < K) {  // stream next K-step tiles through L2
      __builtin_prefetch(gpA + 32, 0, 1);
      __builtin_prefetch(gpB + (size_t)32 * N, 0, 1);
    }
    wait_async0();
    __syncthreads();

    v16h af[2], bf[4];
#pragma unroll
    for (int i = 0; i < 2; ++i) {  // A frag: K(e) = g*8+e | 16+g*8+(e-8)
      const h16* p = &lA[(wm * 32 + i * 16 + hm) * 32];
      af[i] = mk_frag(*(const v8h*)(p + g * 8), *(const v8h*)(p + 16 + g * 8));
    }
#pragma unroll
    for (int j = 0; j < 4; ++j) {  // B frag: K(e) = g*8+e | 16+g*8+(e-8)
      const h16* p = &lBt[(wn * 64 + j * 16 + hm) * 32];
      bf[j] = mk_frag(*(const v8h*)(p + g * 8), *(const v8h*)(p + 16 + g * 8));
    }
#pragma unroll
    for (int i = 0; i < 2; ++i)
#pragma unroll
      for (int j = 0; j < 4; ++j) acc[i][j] = wmma_f16(af[i], bf[j], acc[i][j]);
  }

  // epilogue: D layout m = rr + 8*g, n = hm
#pragma unroll
  for (int i = 0; i < 2; ++i) {
    const int rowBase = r0 + wm * 32 + i * 16 + 8 * g;
#pragma unroll
    for (int j = 0; j < 4; ++j) {
      const int col = c0 + wn * 64 + j * 16 + hm;
      const float bj = bias[col];
#pragma unroll
      for (int rr = 0; rr < 8; ++rr) {
        float v = acc[i][j][rr] + bj;
        size_t off = (size_t)(rowBase + rr) * N + col;
        if (F16OUT) ((h16*)Cout)[off] = (h16)v;
        else        ((float*)Cout)[off] = v;
      }
    }
  }
}

// ---------------------------------------------------------------- qkv scatter
// qkv[b*S+s][c], c = h*192 + t*64 + d  ->  Q/K: [b,h,s,d]; V transposed: [b,h,d,s]
__global__ void scatter_qkv(const h16* __restrict__ qkv, h16* __restrict__ Q,
                            h16* __restrict__ Kh, h16* __restrict__ Vt) {
  int idx = blockIdx.x * 256 + threadIdx.x;
  int c   = idx % (3 * DMODEL);
  int row = idx / (3 * DMODEL);
  int b = row / SEQL, s = row % SEQL;
  int h = c / (3 * HDIM);
  int w = c % (3 * HDIM);
  int t = w / HDIM, d = w % HDIM;
  h16 v = qkv[idx];
  int bh = b * HEADS + h;
  if (t == 0)      Q[((size_t)bh * SEQL + s) * HDIM + d] = v;
  else if (t == 1) Kh[((size_t)bh * SEQL + s) * HDIM + d] = v;
  else             Vt[((size_t)bh * HDIM + d) * SEQL + s] = v;
}

// ---------------------------------------------------------------- flash attention
// One block = (b,h, 128 q-rows); wave owns 16 q-rows. Online softmax over 64-key blocks.
__global__ __launch_bounds__(256) void attn_kernel(
    const h16* __restrict__ Q, const h16* __restrict__ Kh, const h16* __restrict__ Vt,
    const float* __restrict__ mask, h16* __restrict__ vals) {
  __shared__ h16 lP[8 * 16 * 64];  // per-wave 16x64 P tile (D-layout -> A-layout bounce)

  const int tid  = threadIdx.x;
  const int lane = tid & 31;
  const int wid  = tid >> 5;
  const int hm   = lane & 15;
  const int g    = lane >> 4;
  const int QT   = SEQL / 128;
  const int bh   = blockIdx.x / QT;
  const int qt   = blockIdx.x % QT;
  const int b    = bh / HEADS, h = bh % HEADS;
  const int q0   = qt * 128 + wid * 16;  // wave's q-row base within sequence

  const h16* Qbase = Q  + (size_t)bh * SEQL * HDIM;
  const h16* Kbase = Kh + (size_t)bh * SEQL * HDIM;
  const h16* Vbase = Vt + (size_t)bh * HDIM * SEQL;

  v16h qf[2];
#pragma unroll
  for (int kc = 0; kc < 2; ++kc) {  // A frag over d (K dim), row m = q0+hm
    const h16* p = Qbase + (size_t)(q0 + hm) * HDIM + kc * 32;
    qf[kc] = mk_frag(*(const v8h*)(p + g * 8), *(const v8h*)(p + 16 + g * 8));
  }

  float mrun[8], lrun[8];
  v8f oacc[4];
#pragma unroll
  for (int rr = 0; rr < 8; ++rr) { mrun[rr] = -1e30f; lrun[rr] = 0.f; }
#pragma unroll
  for (int j = 0; j < 4; ++j) oacc[j] = {};

  h16* myP = &lP[wid * 16 * 64];

  for (int kb = 0; kb < SEQL; kb += 64) {
    // ---- scores S = Q K^T : 4 n-tiles x 2 k-chunks
    v8f sacc[4];
#pragma unroll
    for (int j = 0; j < 4; ++j) sacc[j] = {};
#pragma unroll
    for (int kc = 0; kc < 2; ++kc) {
#pragma unroll
      for (int j = 0; j < 4; ++j) {
        // B frag column n = key row, K = d contiguous -> two 16B global loads (g*8, 16+g*8)
        const h16* p = Kbase + (size_t)(kb + j * 16 + hm) * HDIM + kc * 32;
        v16h kf = mk_frag(*(const v8h*)(p + g * 8), *(const v8h*)(p + 16 + g * 8));
        sacc[j] = wmma_f16(qf[kc], kf, sacc[j]);
      }
    }
    // ---- scale + additive mask (D layout: row m = rr+8g, col n = hm)
#pragma unroll
    for (int j = 0; j < 4; ++j) {
      const int col = kb + j * 16 + hm;
#pragma unroll
      for (int rr = 0; rr < 8; ++rr) {
        const int row = q0 + rr + 8 * g;
        sacc[j][rr] = sacc[j][rr] * 0.125f + mask[(size_t)row * SEQL + col];
      }
    }
    // ---- row max across 16 key-columns held by the 16-lane group
    float tmax[8];
#pragma unroll
    for (int rr = 0; rr < 8; ++rr) {
      float m = sacc[0][rr];
#pragma unroll
      for (int j = 1; j < 4; ++j) m = fmaxf(m, sacc[j][rr]);
      tmax[rr] = m;
    }
#pragma unroll
    for (int off = 1; off < 16; off <<= 1)
#pragma unroll
      for (int rr = 0; rr < 8; ++rr)
        tmax[rr] = fmaxf(tmax[rr], __shfl_xor(tmax[rr], off, 32));
    // ---- online softmax rescale
    float corr[8];
#pragma unroll
    for (int rr = 0; rr < 8; ++rr) {
      float mn = fmaxf(mrun[rr], tmax[rr]);
      corr[rr] = __expf(mrun[rr] - mn);
      mrun[rr] = mn;
      lrun[rr] *= corr[rr];
    }
#pragma unroll
    for (int j = 0; j < 4; ++j)
#pragma unroll
      for (int rr = 0; rr < 8; ++rr) oacc[j][rr] *= corr[rr];
    // ---- P = exp(S - m), row sums, stash P as f16 in LDS (D-layout write)
    float psum[8];
#pragma unroll
    for (int rr = 0; rr < 8; ++rr) psum[rr] = 0.f;
#pragma unroll
    for (int j = 0; j < 4; ++j)
#pragma unroll
      for (int rr = 0; rr < 8; ++rr) {
        float p = __expf(sacc[j][rr] - mrun[rr]);
        psum[rr] += p;
        myP[(rr + 8 * g) * 64 + j * 16 + hm] = (h16)p;
      }
#pragma unroll
    for (int off = 1; off < 16; off <<= 1)
#pragma unroll
      for (int rr = 0; rr < 8; ++rr) psum[rr] += __shfl_xor(psum[rr], off, 32);
#pragma unroll
    for (int rr = 0; rr < 8; ++rr) lrun[rr] += psum[rr];

    __syncthreads();
    // ---- O += P @ V (P: A frag from LDS; V frags contiguous from Vt)
#pragma unroll
    for (int kc = 0; kc < 2; ++kc) {
      const h16* pp = myP + hm * 64 + kc * 32;
      v16h af = mk_frag(*(const v8h*)(pp + g * 8), *(const v8h*)(pp + 16 + g * 8));
#pragma unroll
      for (int j = 0; j < 4; ++j) {
        const h16* vp = Vbase + (size_t)(j * 16 + hm) * SEQL + kb + kc * 32;
        v16h vf = mk_frag(*(const v8h*)(vp + g * 8), *(const v8h*)(vp + 16 + g * 8));
        oacc[j] = wmma_f16(af, vf, oacc[j]);
      }
    }
    __syncthreads();
  }

  // ---- normalize + store vals[b][s][h*64+d] (f16, feeds out-proj GEMM)
#pragma unroll
  for (int j = 0; j < 4; ++j)
#pragma unroll
    for (int rr = 0; rr < 8; ++rr) {
      const int row = q0 + rr + 8 * g;
      const int d = j * 16 + hm;
      float v = oacc[j][rr] / lrun[rr];
      vals[((size_t)b * SEQL + row) * DMODEL + h * HDIM + d] = (h16)v;
    }
}

// ---------------------------------------------------------------- launch
extern "C" void kernel_launch(void* const* d_in, const int* in_sizes, int n_in,
                              void* d_out, int out_size, void* d_ws, size_t ws_size,
                              hipStream_t stream) {
  (void)in_sizes; (void)n_in; (void)out_size; (void)ws_size;
  const float* x     = (const float*)d_in[0];
  const float* mask  = (const float*)d_in[1];
  const float* w_qkv = (const float*)d_in[2];
  const float* b_qkv = (const float*)d_in[3];
  const float* w_out = (const float*)d_in[4];
  const float* b_out = (const float*)d_in[5];
  float* out = (float*)d_out;

  char* ws = (char*)d_ws;
  size_t off = 0;
  auto alloc = [&](size_t bytes) {
    void* p = ws + off;
    off += (bytes + 255) & ~(size_t)255;
    return p;
  };
  h16* xh    = (h16*)alloc((size_t)BATCH * SEQL * DMODEL * 2);
  h16* wqkvh = (h16*)alloc((size_t)DMODEL * 3 * DMODEL * 2);
  h16* wouth = (h16*)alloc((size_t)DMODEL * DMODEL * 2);
  h16* qkvh  = (h16*)alloc((size_t)BATCH * SEQL * 3 * DMODEL * 2);
  h16* Qh    = (h16*)alloc((size_t)BATCH * HEADS * SEQL * HDIM * 2);
  h16* Kh    = (h16*)alloc((size_t)BATCH * HEADS * SEQL * HDIM * 2);
  h16* Vt    = (h16*)alloc((size_t)BATCH * HEADS * HDIM * SEQL * 2);
  h16* vals  = (h16*)alloc((size_t)BATCH * SEQL * DMODEL * 2);

  const int nx  = BATCH * SEQL * DMODEL;
  const int nw  = DMODEL * 3 * DMODEL;
  const int nwo = DMODEL * DMODEL;
  cast_f32_f16<<<(nx + 255) / 256, 256, 0, stream>>>(x, xh, nx);
  cast_f32_f16<<<(nw + 255) / 256, 256, 0, stream>>>(w_qkv, wqkvh, nw);
  cast_f32_f16<<<(nwo + 255) / 256, 256, 0, stream>>>(w_out, wouth, nwo);

  dim3 g1(3 * DMODEL / 128, BATCH * SEQL / 128);
  gemm_bias<true><<<g1, 256, 0, stream>>>(xh, wqkvh, b_qkv, qkvh,
                                          BATCH * SEQL, 3 * DMODEL, DMODEL);

  const int ns = BATCH * SEQL * 3 * DMODEL;
  scatter_qkv<<<ns / 256, 256, 0, stream>>>(qkvh, Qh, Kh, Vt);

  attn_kernel<<<BATCH * HEADS * (SEQL / 128), 256, 0, stream>>>(Qh, Kh, Vt, mask, vals);

  dim3 g2(DMODEL / 128, BATCH * SEQL / 128);
  gemm_bias<false><<<g2, 256, 0, stream>>>(vals, wouth, b_out, out,
                                           BATCH * SEQL, DMODEL, DMODEL);
}
